// Net_1632087572622
// MI455X (gfx1250) — matrix operand order repaired and checked by reading, not compiled, not run on hardware
//
#include <hip/hip_runtime.h>
#include <hip/hip_bf16.h>
#include <math.h>

// ---------------- problem constants (from reference) ----------------
#define NNODES 8192
#define NEDGES 131072
#define KT     125      // 5^3 spline kernels
#define NCOR   8        // 2^3 B-spline corners per edge

typedef __attribute__((ext_vector_type(16))) __bf16 v16bf;
typedef __attribute__((ext_vector_type(8)))  float  v8f;

// ---------------- spline basis: [E,8] weights + flat kernel ids ----------------
__global__ void k_basis(const float* __restrict__ pseudo,
                        float* __restrict__ basis, int* __restrict__ kidx) {
  int e = blockIdx.x * blockDim.x + threadIdx.x;
  if (e >= NEDGES) return;
  float fr[3]; int lo[3];
#pragma unroll
  for (int d = 0; d < 3; ++d) {
    float s  = pseudo[e * 3 + d] * 4.0f;   // (KS-1) = 4
    float fl = floorf(s);
    lo[d] = (int)fl;
    fr[d] = s - fl;
  }
#pragma unroll
  for (int c = 0; c < NCOR; ++c) {
    float w = 1.0f;
    int k = 0, mul = 1;
#pragma unroll
    for (int d = 0; d < 3; ++d) {
      int b = (c >> d) & 1;
      int v = lo[d] + b;
      v = v < 0 ? 0 : (v > 4 ? 4 : v);
      k += v * mul;
      mul *= 5;
      w *= b ? fr[d] : (1.0f - fr[d]);
    }
    basis[e * NCOR + c] = w;
    kidx [e * NCOR + c] = k;
  }
}

// ---------------- convert f32 -> bf16 (for input features x) ----------------
__global__ void k_f32_to_bf16(const float* __restrict__ src, __bf16* __restrict__ dst, long n) {
  long i = (long)blockIdx.x * blockDim.x + threadIdx.x;
  if (i < n) dst[i] = (__bf16)src[i];
}

// ---------------- pack [W(125,cin,co) ; R(cin,co)] -> BT[co, Ktot] bf16 ----------------
__global__ void k_prep_w(const float* __restrict__ W, const float* __restrict__ R,
                         __bf16* __restrict__ BT, int cin, int co,
                         int Kslot, int Ktot) {
  long idx = (long)blockIdx.x * blockDim.x + threadIdx.x;
  long total = (long)co * Ktot;
  if (idx >= total) return;
  int o = (int)(idx / Ktot);
  int k = (int)(idx % Ktot);
  float v = 0.0f;
  int Kspline = KT * cin;
  if (k < Kspline)                        v = W[(long)k * co + o];            // W[(kt*cin+i), o]
  else if (k >= Kslot && k < Kslot + cin) v = R[(long)(k - Kslot) * co + o];  // root weight rows
  BT[idx] = (__bf16)v;
}

// ---------------- transpose dense weight [K,Nc] f32 -> [Nc,K] bf16 ----------------
__global__ void k_transpose_bf16(const float* __restrict__ Wsrc,
                                 __bf16* __restrict__ BT, int K, int Nc) {
  long idx = (long)blockIdx.x * blockDim.x + threadIdx.x;
  if (idx >= (long)K * Nc) return;
  int o = (int)(idx / K);
  int k = (int)(idx % K);
  BT[idx] = (__bf16)Wsrc[(long)k * Nc + o];
}

// ---------------- copy h into tail K-columns of M (fuses h@R into the big GEMM) ----------------
__global__ void k_tail(const __bf16* __restrict__ h, __bf16* __restrict__ Mbuf,
                       int cin, int Kslot, int Ktot) {
  long idx = (long)blockIdx.x * blockDim.x + threadIdx.x;
  long total = (long)NNODES * cin;
  if (idx >= total) return;
  int n = (int)(idx / cin);
  int i = (int)(idx % cin);
  Mbuf[(long)n * Ktot + Kslot + i] = h[idx];
}

// ---------------- edge scatter into bf16 M via hardware packed-bf16 atomics -------------
// M[dst, kidx*cin + i] += basis * h[src, i]; channel pairs -> one global_atomic_pk_add_bf16.
__global__ void k_scatter_bf16(const __bf16* __restrict__ h, const int* __restrict__ src,
                               const int* __restrict__ dst, const float* __restrict__ basis,
                               const int* __restrict__ kidx, __bf16* __restrict__ Mbuf,
                               int cin, int Ktot, long total) {
  long tid = (long)blockIdx.x * blockDim.x + threadIdx.x;
  if (tid >= total) return;
  int pairs = cin >> 1;                 // 0 when cin==1
  float v0, v1;
  long fbase;
  if (pairs) {
    int  ip = (int)(tid % pairs);
    long ec = tid / pairs;
    int  e  = (int)(ec >> 3);
    float w = basis[ec];
    int   k = kidx[ec];
    int   s = src[e], d = dst[e];
    int   i = ip * 2;
    v0 = w * (float)h[(long)s * cin + i];
    v1 = w * (float)h[(long)s * cin + i + 1];
    fbase = (long)d * Ktot + (long)k * cin + i;     // cin even && i even -> fbase even
  } else {
    long ec = tid;
    int  e  = (int)(ec >> 3);
    float w = basis[ec];
    int   k = kidx[ec];
    int   s = src[e], d = dst[e];
    float val = w * (float)h[s];
    long f = (long)d * Ktot + k;
    if (f & 1) { fbase = f - 1; v0 = 0.0f; v1 = val; }
    else       { fbase = f;     v0 = val;  v1 = 0.0f; }
  }
  __bf16 b0 = (__bf16)v0, b1 = (__bf16)v1;
  unsigned pk = ((unsigned)__builtin_bit_cast(unsigned short, b1) << 16) |
                 (unsigned)__builtin_bit_cast(unsigned short, b0);
  unsigned long long addr = (unsigned long long)(size_t)(Mbuf + fbase);  // 4B aligned
  asm volatile("global_atomic_pk_add_bf16 %0, %1, off"
               :: "v"(addr), "v"(pk) : "memory");
}

// ---------------- WMMA GEMM: C[8192,N] = A[8192,K](bf16) x BT[N,K](bf16)^T + bias --------
// Block = 8 waves = 128x(16*NT) C tile. B tile staged in LDS via async-to-LDS copies,
// double buffered (issue next K-slice, s_wait_asynccnt 1, barrier, consume). A fragments
// are direct b128 loads. Fragment layouts per CDNA5 ISA 7.12.2 (wave32).
template<int NT>
__global__ void __launch_bounds__(256)
k_gemm(const __bf16* __restrict__ A, const __bf16* __restrict__ BT,
       const float* __restrict__ bias, void* __restrict__ Cout,
       int Ncols, int K, int Mblocks, int act, int out_f32) {
  __shared__ __attribute__((aligned(16))) __bf16 tile[2][16 * NT][40]; // 80B row stride: 16B aligned, 20 banks
  int t    = threadIdx.x;
  int wv   = t >> 5;
  int lane = t & 31;
  int lm   = lane & 15;
  int hl   = lane >> 4;
  int bm = blockIdx.x % Mblocks;
  int bn = blockIdx.x / Mblocks;
  int mBase = bm * 128 + wv * 16;
  int nBase = bn * (16 * NT);

  const __bf16* Arow = A + (size_t)(mBase + lm) * K;

  // B-tile loader mapping: thread -> (row in tile, 16B quarter of the 32-elem K slice)
  int rowInTile = t >> 2;           // 0..63
  int quarter   = t & 3;            // 0..3
  bool loader   = (rowInTile < 16 * NT);   // wave-uniform (whole waves idle for NT=2)
  int rEff      = loader ? rowInTile : 0;
  const __bf16* gB = BT + (size_t)(nBase + rEff) * K + quarter * 8;

  v8f acc[NT] = {};

  // prologue: stage first K-slice into buffer 0
  if (loader) {
    unsigned lds = (unsigned)(size_t)&tile[0][rEff][quarter * 8];
    unsigned long long g = (unsigned long long)(size_t)gB;
    asm volatile("global_load_async_to_lds_b128 %0, %1, off"
                 :: "v"(lds), "v"(g) : "memory");
  }

  int buf = 0;
  for (int kk = 0; kk < K; kk += 32, buf ^= 1) {
    bool more = (kk + 32) < K;
    if (more) {                      // issue next slice before waiting on current one
      if (loader) {
        unsigned lds = (unsigned)(size_t)&tile[buf ^ 1][rEff][quarter * 8];
        unsigned long long g = (unsigned long long)(size_t)(gB + kk + 32);
        asm volatile("global_load_async_to_lds_b128 %0, %1, off"
                     :: "v"(lds), "v"(g) : "memory");
      }
      asm volatile("s_wait_asynccnt 0x1" ::: "memory");   // current slice done, next in flight
    } else {
      asm volatile("s_wait_asynccnt 0x0" ::: "memory");
    }
    __syncthreads();                 // LDS slice visible to all waves

    if (kk + 32 < K) __builtin_prefetch(Arow + kk + 32, 0, 0);  // global_prefetch_b8 on A stream

    // ---- A fragment: lane half selects K {0-7,16-23} vs {8-15,24-31} ----
    union { v16bf v; uint4 q[2]; } a;
    a.q[0] = *(const uint4*)(Arow + kk + hl * 8);
    a.q[1] = *(const uint4*)(Arow + kk + 16 + hl * 8);
#pragma unroll
    for (int j = 0; j < NT; ++j) {
      // ---- B fragment from LDS: lane holds column n, K {0-7,16-23} vs {8-15,24-31} by half ----
      union { v16bf v; uint4 q[2]; } b;
      const __bf16* brow = &tile[buf][j * 16 + lm][0];
      b.q[0] = *(const uint4*)(brow + hl * 8);
      b.q[1] = *(const uint4*)(brow + 16 + hl * 8);
      acc[j] = __builtin_amdgcn_wmma_f32_16x16x32_bf16(
                   /*neg_a=*/false, a.v, /*neg_b=*/false, b.v,
                   /*c_mod=*/(short)0, acc[j], /*reuse_a=*/false, /*reuse_b=*/false);
    }
    __syncthreads();                 // all reads done before buffer is overwritten
  }

  // ---- epilogue: bias + optional ELU; store bf16 (activations) or f32 (logits) ----
#pragma unroll
  for (int j = 0; j < NT; ++j) {
    int n = nBase + j * 16 + lm;
    float bv = bias ? bias[n] : 0.0f;
#pragma unroll
    for (int r = 0; r < 8; ++r) {
      int m = mBase + (hl << 3) + r;
      float v = acc[j][r] + bv;
      if (act) v = v > 0.0f ? v : expm1f(v);    // ELU(alpha=1)
      if (out_f32) ((float*)Cout)[(size_t)m * Ncols + n] = v;
      else         ((__bf16*)Cout)[(size_t)m * Ncols + n] = (__bf16)v;
    }
  }
}

// ---------------- row-wise log_softmax (in place), one block per row ----------------
__global__ void k_logsoftmax(float* __restrict__ p, int ncols) {
  __shared__ float red[256];
  float* r = p + (long)blockIdx.x * ncols;
  int tid = threadIdx.x;
  float mx = -3.4e38f;
  for (int j = tid; j < ncols; j += 256) mx = fmaxf(mx, r[j]);
  red[tid] = mx; __syncthreads();
  for (int s = 128; s > 0; s >>= 1) {
    if (tid < s) red[tid] = fmaxf(red[tid], red[tid + s]);
    __syncthreads();
  }
  mx = red[0]; __syncthreads();
  float sum = 0.0f;
  for (int j = tid; j < ncols; j += 256) sum += expf(r[j] - mx);
  red[tid] = sum; __syncthreads();
  for (int s = 128; s > 0; s >>= 1) {
    if (tid < s) red[tid] += red[tid + s];
    __syncthreads();
  }
  float lg = logf(red[0]);
  for (int j = tid; j < ncols; j += 256) r[j] = r[j] - mx - lg;
}

// =====================================================================================
extern "C" void kernel_launch(void* const* d_in, const int* in_sizes, int n_in,
                              void* d_out, int out_size, void* d_ws, size_t ws_size,
                              hipStream_t stream) {
  const float* x      = (const float*)d_in[0];
  const int*   ei     = (const int*)  d_in[1];
  const float* pseudo = (const float*)d_in[2];
  const int* src = ei;
  const int* dst = ei + NEDGES;

  // ---- workspace partition (256B aligned) ----
  char* wp = (char*)d_ws;
  auto alloc = [&](size_t bytes) -> void* {
    void* p = (void*)wp;
    wp += (bytes + 255) & ~(size_t)255;
    return p;
  };
  float*  basis = (float*) alloc((size_t)NEDGES * NCOR * 4);
  int*    kidx  = (int*)   alloc((size_t)NEDGES * NCOR * 4);
  __bf16* hX    = (__bf16*)alloc((size_t)NNODES * 2);           // bf16 copy of x
  __bf16* hA    = (__bf16*)alloc((size_t)NNODES * 256 * 2);
  __bf16* hB    = (__bf16*)alloc((size_t)NNODES * 256 * 2);
  __bf16* BT    = (__bf16*)alloc((size_t)8192 * 256 * 2);       // covers all weight packs
  __bf16* Mbuf  = (__bf16*)alloc((size_t)NNODES * 8064 * 2);    // largest layer Ktot (~132MB, L2-resident)

  k_basis<<<(NEDGES + 255) / 256, 256, 0, stream>>>(pseudo, basis, kidx);
  k_f32_to_bf16<<<(NNODES + 255) / 256, 256, 0, stream>>>(x, hX, NNODES);

  auto gemm = [&](const __bf16* A, const __bf16* BTp, const float* bias,
                  void* C, int Nc, int K, int act, int out_f32) {
    int Mblocks = NNODES / 128;
    if (Nc % 64 == 0) {
      k_gemm<4><<<Mblocks * (Nc / 64), 256, 0, stream>>>(A, BTp, bias, C, Nc, K, Mblocks, act, out_f32);
    } else {
      k_gemm<2><<<Mblocks * (Nc / 32), 256, 0, stream>>>(A, BTp, bias, C, Nc, K, Mblocks, act, out_f32);
    }
  };

  const __bf16* h = hX;
  int cin = 1;
  const int couts[6] = {32, 64, 64, 64, 64, 64};
  for (int li = 0; li < 6; ++li) {
    int co = couts[li];
    const float* W = (const float*)d_in[3 + 3 * li];
    const float* R = (const float*)d_in[4 + 3 * li];
    const float* B = (const float*)d_in[5 + 3 * li];
    int Kslot = ((KT * cin + 31) / 32) * 32;   // spline K, padded to x32
    int tailw = ((cin + 31) / 32) * 32;        // root-weight K, padded to x32
    int Ktot  = Kslot + tailw;                 // 160 / 4032 / 8064 ... all x32

    hipMemsetAsync(Mbuf, 0, (size_t)NNODES * Ktot * 2, stream);
    long wtot = (long)co * Ktot;
    k_prep_w<<<(int)((wtot + 255) / 256), 256, 0, stream>>>(W, R, BT, cin, co, Kslot, Ktot);
    long ttot = (long)NNODES * cin;
    k_tail<<<(int)((ttot + 255) / 256), 256, 0, stream>>>(h, Mbuf, cin, Kslot, Ktot);
    long stot = (long)NEDGES * NCOR * (cin > 1 ? cin / 2 : 1);
    k_scatter_bf16<<<(int)((stot + 255) / 256), 256, 0, stream>>>(h, src, dst, basis, kidx,
                                                                  Mbuf, cin, Ktot, stot);
    __bf16* out = (li & 1) ? hB : hA;          // li=0..5 -> hA,hB,hA,hB,hA,hB
    gemm(Mbuf, BT, B, out, co, Ktot, /*elu=*/1, /*out_f32=*/0);
    h = out;
    cin = co;
  }
  // h == hB after the loop.

  // ---- Linear 64 -> 256, ELU ----
  const float* L1w = (const float*)d_in[21];
  const float* L1b = (const float*)d_in[22];
  k_transpose_bf16<<<(64 * 256 + 255) / 256, 256, 0, stream>>>(L1w, BT, 64, 256);
  gemm(h, BT, L1b, hA, 256, 64, /*elu=*/1, /*out_f32=*/0);

  // ---- Linear 256 -> 8192 straight into d_out (f32 logits) ----
  const float* L2w = (const float*)d_in[23];
  const float* L2b = (const float*)d_in[24];
  long l2tot = (long)256 * 8192;
  k_transpose_bf16<<<(int)((l2tot + 255) / 256), 256, 0, stream>>>(L2w, BT, 256, 8192);
  gemm(hA, BT, L2b, d_out, 8192, 256, /*elu=*/0, /*out_f32=*/1);

  // ---- log_softmax in place on d_out ----
  k_logsoftmax<<<NNODES, 256, 0, stream>>>((float*)d_out, NNODES);
}
